// RNN1_9302899163524
// MI455X (gfx1250) — matrix operand order repaired and checked by reading, not compiled, not run on hardware
//
#include <hip/hip_runtime.h>
#include <hip/hip_bf16.h>

typedef _Float16 v16h __attribute__((ext_vector_type(16)));
typedef _Float16 v8h  __attribute__((ext_vector_type(8)));
typedef float    v8f  __attribute__((ext_vector_type(8)));
typedef float    v4f  __attribute__((ext_vector_type(4)));

// Problem sizes
#define Bb    256
#define Tt    128
#define Ff    1024
#define Hh    400
#define Oo    3
#define HpK   416           // K-padding of H for recurrence: 13*32
#define HpN   448           // N-padding of H (storage stride): 28*16, 7 groups of 4 tiles
#define NTile 28            // HpN/16
#define NG    7             // NTile/4  (4-way N blocking)
#define KT_WH 32            // Ff/32
#define KT_WR 13            // HpK/32

// ---------------------------------------------------------------------------
// Pack row-major fp32 W[K x N] into WMMA-B f16 fragment layout:
// [ntile][ktile][lane(32)][16 halves]; lane L: n = nt*16+(L&15),
// K slab = kt*32 + (L>>4)*8 + (j<8 ? j : j+8). OOB K/N packed as 0 (padding semantics).
// ---------------------------------------------------------------------------
__global__ void pack_b_kernel(const float* __restrict__ W, _Float16* __restrict__ P,
                              int K, int N, int KT, int NTiles) {
  int gid = blockIdx.x * blockDim.x + threadIdx.x;
  int total = NTiles * KT * 32;
  if (gid >= total) return;
  int lane = gid & 31;
  int kt   = (gid >> 5) % KT;
  int nt   = (gid >> 5) / KT;
  int n  = nt * 16 + (lane & 15);
  int kb = kt * 32 + (lane >> 4) * 8;
  _Float16* dst = P + (size_t)gid * 16;
#pragma unroll
  for (int j = 0; j < 16; ++j) {
    int k = kb + (j < 8 ? j : j + 8);
    float v = (k < K && n < N) ? W[(size_t)k * N + n] : 0.0f;
    dst[j] = (_Float16)v;
  }
}

// A-fragment from fp32 row (converted):
// halves 0..7 = A[row][kb+0..7] ; halves 8..15 = A[row][kb+16..23]
__device__ inline v16h load_cvt_a(const float* __restrict__ p) {
  v4f a0 = *(const v4f*)(p);
  v4f a1 = *(const v4f*)(p + 4);
  v4f a2 = *(const v4f*)(p + 16);
  v4f a3 = *(const v4f*)(p + 20);
  v16h a;
  a[0]  = (_Float16)a0[0]; a[1]  = (_Float16)a0[1];
  a[2]  = (_Float16)a0[2]; a[3]  = (_Float16)a0[3];
  a[4]  = (_Float16)a1[0]; a[5]  = (_Float16)a1[1];
  a[6]  = (_Float16)a1[2]; a[7]  = (_Float16)a1[3];
  a[8]  = (_Float16)a2[0]; a[9]  = (_Float16)a2[1];
  a[10] = (_Float16)a2[2]; a[11] = (_Float16)a2[3];
  a[12] = (_Float16)a3[0]; a[13] = (_Float16)a3[1];
  a[14] = (_Float16)a3[2]; a[15] = (_Float16)a3[3];
  return a;
}

// A-fragment from row-major f16 storage: two contiguous 16B loads, no cvt.
__device__ inline v16h load_a_f16(const _Float16* __restrict__ p) {
  v8h lo = *(const v8h*)(p);
  v8h hi = *(const v8h*)(p + 16);
  v16h a;
#pragma unroll
  for (int i = 0; i < 8; ++i) { a[i] = lo[i]; a[8 + i] = hi[i]; }
  return a;
}

// Load the 4 B fragments (one per N-tile in the group) for K-step kk.
template <int KT>
__device__ inline void load_b4(const _Float16* __restrict__ bbase, int kk,
                               v16h& b0, v16h& b1, v16h& b2, v16h& b3) {
  b0 = *(const v16h*)(bbase + (size_t)kk * 512);
  b1 = *(const v16h*)(bbase + (size_t)(1 * KT * 512) + (size_t)kk * 512);
  b2 = *(const v16h*)(bbase + (size_t)(2 * KT * 512) + (size_t)kk * 512);
  b3 = *(const v16h*)(bbase + (size_t)(3 * KT * 512) + (size_t)kk * 512);
}

#define WMMA4(A, B0, B1, B2, B3)                                                            \
  c0 = __builtin_amdgcn_wmma_f32_16x16x32_f16(false, A, false, B0, (short)0, c0, false, false); \
  c1 = __builtin_amdgcn_wmma_f32_16x16x32_f16(false, A, false, B1, (short)0, c1, false, false); \
  c2 = __builtin_amdgcn_wmma_f32_16x16x32_f16(false, A, false, B2, (short)0, c2, false, false); \
  c3 = __builtin_amdgcn_wmma_f32_16x16x32_f16(false, A, false, B3, (short)0, c3, false, false);

// ---------------------------------------------------------------------------
// Phase 1: xin[32768 x HpN](f16) = x[32768 x Ff](f32) * Wh + bh
// One wave per 16x64 strip (4 N-tiles). Software-pipelined: loads for step
// kk+1 are issued before the 4 WMMAs of step kk.
// ---------------------------------------------------------------------------
__global__ __launch_bounds__(128) void gemm_xin_kernel(
    const float* __restrict__ x, const _Float16* __restrict__ WhP,
    const float* __restrict__ bh, _Float16* __restrict__ xin) {
  int wave = threadIdx.x >> 5;
  int lane = threadIdx.x & 31;
  int wid  = blockIdx.x * 4 + wave;     // 2048*7 waves, exact
  int ntg  = wid % NG;
  int mt   = wid / NG;
  int hi   = lane >> 4;
  int mrow = mt * 16 + (lane & 15);
  const float* arow = x + (size_t)mrow * Ff + hi * 8;
  const _Float16* bbase = WhP + ((size_t)(ntg * 4 * KT_WH) * 32 + lane) * 16;
  v8f c0 = {}, c1 = {}, c2 = {}, c3 = {};

  v16h a = load_cvt_a(arow);
  v16h b0, b1, b2, b3;
  load_b4<KT_WH>(bbase, 0, b0, b1, b2, b3);
#pragma unroll 2
  for (int kk = 0; kk < KT_WH - 1; ++kk) {
    v16h an = load_cvt_a(arow + (kk + 1) * 32);
    v16h n0, n1, n2, n3;
    load_b4<KT_WH>(bbase, kk + 1, n0, n1, n2, n3);
    WMMA4(a, b0, b1, b2, b3)
    a = an; b0 = n0; b1 = n1; b2 = n2; b3 = n3;
  }
  WMMA4(a, b0, b1, b2, b3)

  int ncol = lane & 15;
  int r0   = mt * 16 + hi * 8;
#define STORE_TILE(J, CJ)                                                     \
  {                                                                           \
    int n = (ntg * 4 + (J)) * 16 + ncol;                                      \
    float bias = (n < Hh) ? bh[n] : 0.0f;                                     \
    _Float16* optr = xin + (size_t)r0 * HpN + n;                              \
    _Pragma("unroll") for (int r = 0; r < 8; ++r)                             \
        optr[(size_t)r * HpN] = (_Float16)(CJ[r] + bias);                     \
  }
  STORE_TILE(0, c0) STORE_TILE(1, c1) STORE_TILE(2, c2) STORE_TILE(3, c3)
#undef STORE_TILE
}

// ---------------------------------------------------------------------------
// Phase 2 step: hnext(f16)[Bb x HpN] = sigmoid(xin[:,t,:] + hprev(f16) * Wr)
// 16 M-tiles x 7 N-groups = 112 waves; K loop 13 steps, pipelined, no cvt.
// ---------------------------------------------------------------------------
__global__ __launch_bounds__(128) void rnn_step_kernel(
    const _Float16* __restrict__ hprev, const _Float16* __restrict__ WrP,
    const _Float16* __restrict__ xin, int t, _Float16* __restrict__ hnext) {
  int wave = threadIdx.x >> 5;
  int lane = threadIdx.x & 31;
  int wid  = blockIdx.x * 4 + wave;     // 16*7 = 112 waves, exact
  int ntg  = wid % NG;
  int mt   = wid / NG;
  int hi   = lane >> 4;
  int brow = mt * 16 + (lane & 15);
  const _Float16* arow  = hprev + (size_t)brow * HpN + hi * 8;
  const _Float16* bbase = WrP + ((size_t)(ntg * 4 * KT_WR) * 32 + lane) * 16;
  v8f c0 = {}, c1 = {}, c2 = {}, c3 = {};

  v16h a = load_a_f16(arow);
  v16h b0, b1, b2, b3;
  load_b4<KT_WR>(bbase, 0, b0, b1, b2, b3);
#pragma unroll 2
  for (int kk = 0; kk < KT_WR - 1; ++kk) {
    v16h an = load_a_f16(arow + (kk + 1) * 32);
    v16h n0, n1, n2, n3;
    load_b4<KT_WR>(bbase, kk + 1, n0, n1, n2, n3);
    WMMA4(a, b0, b1, b2, b3)
    a = an; b0 = n0; b1 = n1; b2 = n2; b3 = n3;
  }
  WMMA4(a, b0, b1, b2, b3)

  int ncol = lane & 15;
  int b0r  = mt * 16 + hi * 8;
#define STORE_TILE(J, CJ)                                                     \
  {                                                                           \
    int n = (ntg * 4 + (J)) * 16 + ncol;                                      \
    _Pragma("unroll") for (int r = 0; r < 8; ++r) {                           \
      int bb = b0r + r;                                                       \
      float z = CJ[r] + (float)xin[((size_t)bb * Tt + t) * HpN + n];          \
      hnext[(size_t)bb * HpN + n] = (_Float16)(1.0f / (1.0f + __expf(-z)));   \
    }                                                                         \
  }
  STORE_TILE(0, c0) STORE_TILE(1, c1) STORE_TILE(2, c2) STORE_TILE(3, c3)
#undef STORE_TILE
}

// Zero-init h0 (f16 buffer) as 32-bit words.
__global__ void zero_kernel(unsigned int* __restrict__ p, int n) {
  int i = blockIdx.x * blockDim.x + threadIdx.x;
  if (i < n) p[i] = 0u;
}

// Phase 3: out[Bb x 3](f32) = sigmoid(h_last * Wo + bo). Tiny; fp32 VALU.
__global__ void out_kernel(const _Float16* __restrict__ h, const float* __restrict__ Wo,
                           const float* __restrict__ bo, float* __restrict__ out) {
  int b = blockIdx.x * blockDim.x + threadIdx.x;
  if (b >= Bb) return;
  float acc0 = bo[0], acc1 = bo[1], acc2 = bo[2];
  const _Float16* hr = h + (size_t)b * HpN;
  for (int k = 0; k < Hh; ++k) {
    float hv = (float)hr[k];
    acc0 += hv * Wo[k * 3 + 0];
    acc1 += hv * Wo[k * 3 + 1];
    acc2 += hv * Wo[k * 3 + 2];
  }
  out[b * 3 + 0] = 1.0f / (1.0f + __expf(-acc0));
  out[b * 3 + 1] = 1.0f / (1.0f + __expf(-acc1));
  out[b * 3 + 2] = 1.0f / (1.0f + __expf(-acc2));
}

extern "C" void kernel_launch(void* const* d_in, const int* in_sizes, int n_in,
                              void* d_out, int out_size, void* d_ws, size_t ws_size,
                              hipStream_t stream) {
  const float* x  = (const float*)d_in[0];   // [B,T,F]
  const float* Wh = (const float*)d_in[1];   // [F,H]
  const float* Wr = (const float*)d_in[2];   // [H,H]
  const float* bh = (const float*)d_in[3];   // [H]
  const float* Wo = (const float*)d_in[4];   // [H,O]
  const float* bo = (const float*)d_in[5];   // [O]
  float* out = (float*)d_out;                // [B,3] fp32

  // Workspace layout (bytes, 256-aligned):
  char* ws = (char*)d_ws;
  _Float16* WhP = (_Float16*)(ws + 0);              //    917,504 B (28*32*512 halves)
  _Float16* WrP = (_Float16*)(ws + 917504);         //    372,736 B (28*13*512 halves)
  _Float16* xin = (_Float16*)(ws + 1290240);        // 29,360,128 B (32768*448 halves)
  _Float16* hA  = (_Float16*)(ws + 30650368);       //    229,376 B (256*448 halves)
  _Float16* hB  = (_Float16*)(ws + 30879744);       //    229,376 B
  // total: 31,109,120 B (~29.7 MB)

  // Pack weights into WMMA-B f16 fragment layout (zero padded).
  pack_b_kernel<<<112, 256, 0, stream>>>(Wh, WhP, Ff, Hh, KT_WH, NTile); // 28672 thr
  pack_b_kernel<<<46, 256, 0, stream>>>(Wr, WrP, Hh, Hh, KT_WR, NTile);  // 11648 thr

  // h0 = 0 (f16 zeros == 0x0000)
  zero_kernel<<<(Bb * HpN / 2) / 256, 256, 0, stream>>>((unsigned int*)hA, Bb * HpN / 2);

  // Phase 1: big input-projection GEMM (14336 waves / 3584 blocks).
  gemm_xin_kernel<<<(2048 * NG) / 4, 128, 0, stream>>>(x, WhP, bh, xin);

  // Phase 2: 128 sequential recurrence steps (ping-pong h).
  _Float16* hcur = hA;
  _Float16* hnxt = hB;
  for (int t = 0; t < Tt; ++t) {
    rnn_step_kernel<<<(16 * NG) / 4, 128, 0, stream>>>(hcur, WrP, xin, t, hnxt);
    _Float16* tmp = hcur; hcur = hnxt; hnxt = tmp;
  }

  // Phase 3: output head.
  out_kernel<<<1, 256, 0, stream>>>(hcur, Wo, bo, out);
}
